// SpatialTransformerBlock_17849884082718
// MI455X (gfx1250) — matrix operand order repaired and checked by reading, not compiled, not run on hardware
//
#include <hip/hip_runtime.h>
#include <hip/hip_bf16.h>

#define WSZ 8
#define SHIFT 4
#define HEADS 4
#define CDIM 64
#define HDIM 256
#define IMGH 256
#define IMGW 256
#define BATCH 4

typedef __bf16 v16bf __attribute__((ext_vector_type(16)));
typedef float v8f __attribute__((ext_vector_type(8)));
typedef float f32x4 __attribute__((ext_vector_type(4)));
typedef unsigned int u32x4 __attribute__((ext_vector_type(4)));
typedef unsigned int u32x8 __attribute__((ext_vector_type(8)));

union FragCast { u32x4 q[2]; v16bf v; };

// A fragment (16-bit A 16x32): row = lane&15, K = kb..kb+7 and kb+16..kb+23, kb = 8*(lane>>4)
__device__ __forceinline__ v16bf frag_a(const __bf16* m, int r0, int ld, int k0, int lane) {
  const __bf16* p = m + (size_t)(r0 + (lane & 15)) * ld + k0 + ((lane >> 4) << 3);
  FragCast u;
  u.q[0] = *(const u32x4*)p;
  u.q[1] = *(const u32x4*)(p + 16);
  return u.v;
}
// B fragment from N x K row-major (B^T == the (out,in) weight matrix):
// col n = lane&15, K = kb..kb+7 and kb+16..kb+23, kb = k0 + 8*(lane>>4)
__device__ __forceinline__ v16bf frag_b(const __bf16* m, int n0, int ld, int k0, int lane) {
  const __bf16* p = m + (size_t)(n0 + (lane & 15)) * ld + k0 + ((lane >> 4) << 3);
  FragCast u;
  u.q[0] = *(const u32x4*)p;
  u.q[1] = *(const u32x4*)(p + 16);
  return u.v;
}

__device__ __forceinline__ v8f wmma_bf16(v16bf a, v16bf b, v8f c) {
  return __builtin_amdgcn_wmma_f32_16x16x32_bf16(false, a, false, b, (short)0, c, false, false);
}

__device__ __forceinline__ float gelu_f(float v) {
  return 0.5f * v * (1.0f + erff(v * 0.70710678118f));
}

// ---- gfx1250 async global->LDS (per-lane, ASYNCcnt) ----
__device__ __forceinline__ unsigned int lds_addr32(const void* p) {
  return (unsigned int)(unsigned long long)p;   // generic LDS addr: offset in low 32 bits
}
__device__ __forceinline__ void async_ld_b32(void* lds_dst, const void* gbase, unsigned int goff) {
  asm volatile("global_load_async_to_lds_b32 %0, %1, %2"
               :: "v"(lds_addr32(lds_dst)), "v"(goff), "s"(gbase)
               : "memory");
}
__device__ __forceinline__ void wait_async0() {
  asm volatile("s_wait_asynccnt 0x0" ::: "memory");
}

// ---- gfx1250 Tensor Data Mover: 1-D block copy global -> LDS (TENSORcnt) ----
// D# per CDNA5 ISA: group0 = {count=1, lds_addr,
// global_addr[56:0], type=2}; group1 = {data_size=8B, tensor_dim0=n8, tile_dim0=n8}.
__device__ __forceinline__ void tdm_copy_1d(void* lds_dst, const void* gsrc, unsigned int n8) {
  const unsigned long long ga = (unsigned long long)gsrc;
  u32x4 g0;
  g0[0] = 1u;                                        // count=1, user mode
  g0[1] = lds_addr32(lds_dst);                       // lds_addr (bytes)
  g0[2] = (unsigned int)ga;                          // global_addr[31:0]
  g0[3] = (unsigned int)(ga >> 32) | (2u << 30);     // global_addr[56:32] | type=2
  u32x8 g1;
  g1[0] = 3u << 16;                                  // workgroup_mask=0, data_size=3 (8B)
  g1[1] = (n8 & 0xFFFFu) << 16;                      // tensor_dim0[15:0] @ bits 63:48
  g1[2] = ((n8 >> 16) & 0xFFFFu) | (1u << 16);       // tensor_dim0[31:16], tensor_dim1=1
  g1[3] = (n8 & 0xFFFFu) << 16;                      // tile_dim0 @ bits 127:112
  g1[4] = 0u;                                        // tile_dim1=0 (unused), tile_dim2=0
  g1[5] = n8;                                        // tensor_dim0_stride[31:0]
  g1[6] = 0u;
  g1[7] = 0u;
  asm volatile("tensor_load_to_lds %0, %1" :: "s"(g0), "s"(g1) : "memory");
}

// ---------------- Kernel 0: one-shot f32->bf16 conversion of all GEMM weights ----------------
// wbf layout: [qkvw 12288][projw 4096][fc1w 16384][fc2w 16384]
__global__ __launch_bounds__(256) void wcvt_kernel(
    const float* __restrict__ qkvw, const float* __restrict__ projw,
    const float* __restrict__ fc1w, const float* __restrict__ fc2w,
    __bf16* __restrict__ wbf)
{
  const int i = blockIdx.x * 256 + threadIdx.x;
  const int total = 12288 + 4096 + 16384 + 16384;
  if (i >= total) return;
  float v;
  if (i < 12288)            v = qkvw[i];
  else if (i < 16384)       v = projw[i - 12288];
  else if (i < 32768)       v = fc1w[i - 16384];
  else                      v = fc2w[i - 32768];
  wbf[i] = (__bf16)v;
}

// ---------------- Kernel 1: LN1 + shifted-window attention + proj + residual ----------------
struct __align__(16) K1U1S { __bf16 qkvw[192 * 64]; __bf16 xw[64 * 64]; };
union  __align__(16) K1U1  { K1U1S s; float S[HEADS][64][64]; };
struct __align__(16) K1U2S { __bf16 q[HEADS][64][32]; __bf16 k[HEADS][64][32]; };
union  __align__(16) K1U2  { K1U2S s; __bf16 P[HEADS][64][64]; float fo[64 * 65]; };

__global__ __launch_bounds__(128) void attn_kernel(
    const float* __restrict__ x, const float* __restrict__ n1w, const float* __restrict__ n1b,
    const __bf16* __restrict__ qkvw_bf, const float* __restrict__ qkvb,
    const float* __restrict__ rpb, const __bf16* __restrict__ projw_bf,
    const float* __restrict__ projb, float* __restrict__ x3)
{
  __shared__ K1U1 u1;
  __shared__ K1U2 u2;
  __shared__ __align__(16) __bf16 vT[HEADS][16][64];
  __shared__ __align__(16) __bf16 ot[64][64];
  __shared__ __align__(16) __bf16 pw[64 * 64];
  __shared__ __align__(16) float xt[64 * 65];     // raw x window tile (LN input + residual)
  __shared__ float qb[192];
  __shared__ float pb[64];
  __shared__ float rp[225 * HEADS];
  __shared__ int  rgn[64];

  const int tid = threadIdx.x;
  const int lane = tid & 31;
  const int wvu = __builtin_amdgcn_readfirstlane(tid >> 5);  // SGPR wave id
  const int wid = blockIdx.x;
  const int bimg = wid >> 10;
  const int wy = (wid & 1023) >> 5, wx = wid & 31;
  const int HW = IMGH * IMGW;
  const int col = lane & 15;
  const int rhalf = (lane >> 4) << 3;

  // TDM-stage bf16 weights (wave 0 issues; TENSORcnt)
  if (wvu == 0) {
    tdm_copy_1d(u1.s.qkvw, qkvw_bf, (192 * 64 * 2) / 8);
    tdm_copy_1d(pw, projw_bf, (64 * 64 * 2) / 8);
  }

  // async per-lane gather of the raw x window tile into LDS (stride-65 rows)
  for (int e = tid; e < 4096; e += 128) {
    const int c = e >> 6, t = e & 63;
    const int ty = t >> 3, tx = t & 7;
    const int sy = (wy * WSZ + ty + SHIFT) & (IMGH - 1);
    const int sx = (wx * WSZ + tx + SHIFT) & (IMGW - 1);
    async_ld_b32(&xt[t * 65 + c], x,
                 (unsigned int)(((bimg * CDIM + c) * HW + sy * IMGW + sx) * 4));
  }
  for (int i = tid; i < 192; i += 128) qb[i] = qkvb[i];
  if (tid < 64) pb[tid] = projb[tid];
  for (int i = tid; i < 225 * HEADS; i += 128) rp[i] = rpb[i];
  { unsigned int* z = (unsigned int*)&u2;
    for (int i = tid; i < (int)(sizeof(K1U2S) / 4); i += 128) z[i] = 0u; }

  if (wvu == 0) __builtin_amdgcn_s_wait_tensorcnt(0);
  wait_async0();
  __syncthreads();

  // LN1 per token from the LDS tile
  if (tid < 64) {
    const int t = tid, ty = t >> 3, tx = t & 7;
    const int gy = wy * WSZ + ty, gx = wx * WSZ + tx;
    rgn[t] = (gy < IMGH - WSZ ? 0 : (gy < IMGH - SHIFT ? 1 : 2)) * 3
           + (gx < IMGW - WSZ ? 0 : (gx < IMGW - SHIFT ? 1 : 2));
    float mu = 0.f;
    for (int c = 0; c < CDIM; ++c) mu += xt[t * 65 + c];
    mu *= (1.f / CDIM);
    float var = 0.f;
    for (int c = 0; c < CDIM; ++c) { float d = xt[t * 65 + c] - mu; var += d * d; }
    const float rs = rsqrtf(var * (1.f / CDIM) + 1e-5f);
    unsigned int* xwrow = (unsigned int*)(u1.s.xw + t * CDIM);
    for (int c = 0; c < CDIM; c += 2) {
      union { __bf16 h[2]; unsigned int u; } pk;
      pk.h[0] = (__bf16)((xt[t * 65 + c]     - mu) * rs * n1w[c]     + n1b[c]);
      pk.h[1] = (__bf16)((xt[t * 65 + c + 1] - mu) * rs * n1w[c + 1] + n1b[c + 1]);
      xwrow[c >> 1] = pk.u;
    }
  }
  __syncthreads();

  // qkv GEMM: (64x64) @ (64x192); scatter branch is scalar (nt derived from SGPR wave id)
  for (int ntl = 0; ntl < 3; ++ntl) {
    const int nt = wvu * 3 + ntl;
    for (int mt = 0; mt < 4; ++mt) {
      v8f acc = {};
      for (int ks = 0; ks < 2; ++ks)
        acc = wmma_bf16(frag_a(u1.s.xw, mt * 16, 64, ks * 32, lane),
                        frag_b(u1.s.qkvw, nt * 16, 64, ks * 32, lane), acc);
      const float bias = qb[nt * 16 + col];
      const int rb = mt * 16 + rhalf;
      if (nt < 4) {
        for (int j = 0; j < 8; ++j)
          u2.s.q[nt][rb + j][col] = (__bf16)((acc[j] + bias) * 0.25f);
      } else if (nt < 8) {
        for (int j = 0; j < 8; ++j)
          u2.s.k[nt - 4][rb + j][col] = (__bf16)(acc[j] + bias);
      } else {
        for (int j = 0; j < 8; ++j)
          vT[nt - 8][col][rb + j] = (__bf16)(acc[j] + bias);
      }
    }
  }
  __syncthreads();

  // S = q @ k^T (+ rel-pos bias + shift mask); one head per wave
  const int h = wvu;
  for (int mt = 0; mt < 4; ++mt)
    for (int nt = 0; nt < 4; ++nt) {
      v8f acc = {};
      acc = wmma_bf16(frag_a(&u2.s.q[h][0][0], mt * 16, 32, 0, lane),
                      frag_b(&u2.s.k[h][0][0], nt * 16, 32, 0, lane), acc);
      const int n = nt * 16 + col;
      const int jy = n >> 3, jx = n & 7, rn = rgn[n];
      const int rb = mt * 16 + rhalf;
      for (int j = 0; j < 8; ++j) {
        const int m = rb + j, iy = m >> 3, ix = m & 7;
        const float bias = rp[((iy - jy + 7) * 15 + (ix - jx + 7)) * HEADS + h];
        const float msk = (rgn[m] != rn) ? -100.f : 0.f;
        u1.S[h][m][n] = acc[j] + bias + msk;
      }
    }
  __syncthreads();   // all waves done with q/k before P overwrites them

  // softmax (2 rows per lane), packed b32 stores of P
  for (int rr = 0; rr < 2; ++rr) {
    const int m = rr * 32 + lane;
    float mx = -1e30f;
    for (int n = 0; n < 64; ++n) mx = fmaxf(mx, u1.S[h][m][n]);
    float s = 0.f;
    for (int n = 0; n < 64; ++n) s += __expf(u1.S[h][m][n] - mx);
    const float inv = 1.f / s;
    unsigned int* prow = (unsigned int*)&u2.P[h][m][0];
    for (int n = 0; n < 64; n += 2) {
      union { __bf16 hh[2]; unsigned int u; } pk;
      pk.hh[0] = (__bf16)(__expf(u1.S[h][m][n]     - mx) * inv);
      pk.hh[1] = (__bf16)(__expf(u1.S[h][m][n + 1] - mx) * inv);
      prow[n >> 1] = pk.u;
    }
  }

  // out = P @ V  (B^T = vT, 16 rows of d)
  for (int mt = 0; mt < 4; ++mt) {
    v8f acc = {};
    for (int ks = 0; ks < 2; ++ks)
      acc = wmma_bf16(frag_a(&u2.P[h][0][0], mt * 16, 64, ks * 32, lane),
                      frag_b(&vT[h][0][0], 0, 64, ks * 32, lane), acc);
    const int rb = mt * 16 + rhalf;
    for (int j = 0; j < 8; ++j) ot[rb + j][h * 16 + col] = (__bf16)acc[j];
  }
  __syncthreads();   // ot ready; P dead -> fo may overwrite

  // proj GEMM -> f32 staging (u2.fo), then coalesced residual writes
  {
    const int nt = wvu;
    for (int mt = 0; mt < 4; ++mt) {
      v8f acc = {};
      for (int ks = 0; ks < 2; ++ks)
        acc = wmma_bf16(frag_a(&ot[0][0], mt * 16, 64, ks * 32, lane),
                        frag_b(pw, nt * 16, 64, ks * 32, lane), acc);
      const int n = nt * 16 + col;
      const float bias = pb[n];
      const int rb = mt * 16 + rhalf;
      for (int j = 0; j < 8; ++j) u2.fo[(rb + j) * 65 + n] = acc[j] + bias;
    }
  }
  __syncthreads();

  if (tid < 64) {
    const int t = tid, ty = t >> 3, tx = t & 7;
    const int sy = (wy * WSZ + ty + SHIFT) & (IMGH - 1);
    const int sx = (wx * WSZ + tx + SHIFT) & (IMGW - 1);
    float* dst = x3 + ((size_t)bimg * HW + (size_t)sy * IMGW + sx) * CDIM;
    for (int c = 0; c < CDIM; c += 4) {
      f32x4 v;
      v.x = xt[t * 65 + c]     + u2.fo[t * 65 + c];
      v.y = xt[t * 65 + c + 1] + u2.fo[t * 65 + c + 1];
      v.z = xt[t * 65 + c + 2] + u2.fo[t * 65 + c + 2];
      v.w = xt[t * 65 + c + 3] + u2.fo[t * 65 + c + 3];
      *(f32x4*)(dst + c) = v;
    }
  }
}

// ---------------- Kernel 2: LN2 + fc1 + GELU -> hid (bf16), 4 token tiles per block ----------------
__global__ __launch_bounds__(128) void ffn1_kernel(
    const float* __restrict__ x3, const float* __restrict__ n2w, const float* __restrict__ n2b,
    const __bf16* __restrict__ fc1w_bf, const float* __restrict__ fc1b, __bf16* __restrict__ hid)
{
  __shared__ __align__(16) __bf16 xw[64 * 64];
  __shared__ __align__(16) __bf16 w1[HDIM * 64];
  __shared__ __align__(16) __bf16 ho[64 * HDIM];
  __shared__ float b1[HDIM];
  const int tid = threadIdx.x, lane = tid & 31;
  const int wvu = __builtin_amdgcn_readfirstlane(tid >> 5);
  const int col = lane & 15, rhalf = (lane >> 4) << 3;
  const size_t tok0 = (size_t)blockIdx.x * 256;

  if (wvu == 0) tdm_copy_1d(w1, fc1w_bf, (HDIM * 64 * 2) / 8);
  for (int i = tid; i < HDIM; i += 128) b1[i] = fc1b[i];
  if (wvu == 0) __builtin_amdgcn_s_wait_tensorcnt(0);

  for (int tile = 0; tile < 4; ++tile) {
    const size_t tok0t = tok0 + tile * 64;
    if (tid < 64) {
      const float* p = x3 + (tok0t + tid) * CDIM;
      float mu = 0.f;
      for (int c = 0; c < CDIM; ++c) mu += p[c];
      mu *= (1.f / CDIM);
      float var = 0.f;
      for (int c = 0; c < CDIM; ++c) { float d = p[c] - mu; var += d * d; }
      const float rs = rsqrtf(var * (1.f / CDIM) + 1e-5f);
      unsigned int* xwrow = (unsigned int*)(xw + tid * CDIM);
      for (int c = 0; c < CDIM; c += 2) {
        union { __bf16 h[2]; unsigned int u; } pk;
        pk.h[0] = (__bf16)((p[c]     - mu) * rs * n2w[c]     + n2b[c]);
        pk.h[1] = (__bf16)((p[c + 1] - mu) * rs * n2w[c + 1] + n2b[c + 1]);
        xwrow[c >> 1] = pk.u;
      }
    }
    __syncthreads();

    for (int ntl = 0; ntl < 4; ++ntl) {
      const int nt = wvu * 4 + ntl;
      for (int mt = 0; mt < 4; ++mt) {
        v8f acc = {};
        for (int ks = 0; ks < 2; ++ks)
          acc = wmma_bf16(frag_a(xw, mt * 16, 64, ks * 32, lane),
                          frag_b(w1, nt * 16, 64, ks * 32, lane), acc);
        const int n = nt * 16 + col;
        const float bb = b1[n];
        const int rb = mt * 16 + rhalf;
        for (int j = 0; j < 8; ++j)
          ho[(rb + j) * HDIM + n] = (__bf16)gelu_f(acc[j] + bb);
      }
    }
    __syncthreads();

    // coalesced copy-out: 64 tokens * 256 ch contiguous
    {
      u32x4* dst = (u32x4*)(hid + tok0t * HDIM);
      const u32x4* src = (const u32x4*)ho;
      for (int i = tid; i < 2048; i += 128) dst[i] = src[i];
    }
    __syncthreads();
  }
}

// ---------------- Kernel 3: dw-conv 5x5 + GELU + add + fc2 + residual -> out (b,c,h,w) ----------------
__global__ __launch_bounds__(128) void ffn2_kernel(
    const __bf16* __restrict__ hid, const float* __restrict__ dww, const float* __restrict__ dwb,
    const __bf16* __restrict__ fc2w_bf, const float* __restrict__ fc2b,
    const float* __restrict__ x3, float* __restrict__ out)
{
  __shared__ __align__(16) __bf16 nb[5][68][64];
  __shared__ __align__(16) __bf16 A[64][HDIM];
  __shared__ __align__(16) __bf16 w2[64 * HDIM];
  __shared__ float dwc[HDIM * 25];
  __shared__ float b2[64];
  const int tid = threadIdx.x, lane = tid & 31;
  const int wvu = __builtin_amdgcn_readfirstlane(tid >> 5);
  const int col = lane & 15, rhalf = (lane >> 4) << 3;
  const int bimg = blockIdx.x >> 8;
  const int y = blockIdx.x & 255;
  const int HW = IMGH * IMGW;
  const __bf16* hb = hid + (size_t)bimg * HW * HDIM;

  if (wvu == 0) tdm_copy_1d(w2, fc2w_bf, (64 * HDIM * 2) / 8);
  for (int i = tid; i < HDIM * 25; i += 128) dwc[i] = dww[i];
  if (tid < 64) b2[tid] = fc2b[tid];
  if (wvu == 0) __builtin_amdgcn_s_wait_tensorcnt(0);

  for (int x0 = 0; x0 < IMGW; x0 += 64) {
    for (int cc = 0; cc < 4; ++cc) {
      const int ch0 = cc * 64;
      for (int e = tid; e < 5 * 68 * 32; e += 128) {
        const int c2 = e & 31, xx = (e >> 5) % 68, r = e / (32 * 68);
        const int yy = y + r - 2, gx2 = x0 + xx - 2;
        unsigned int val = 0u;
        if (yy >= 0 && yy < IMGH && gx2 >= 0 && gx2 < IMGW)
          val = *(const unsigned int*)(hb + ((size_t)yy * IMGW + gx2) * HDIM + ch0 + c2 * 2);
        *(unsigned int*)&nb[r][xx][c2 * 2] = val;
      }
      __syncthreads();
      for (int o = tid; o < 4096; o += 128) {
        const int ch = o & 63, tk = o >> 6;
        float acc = dwb[ch0 + ch];
        for (int r = 0; r < 5; ++r)
          for (int s = 0; s < 5; ++s)
            acc += dwc[(ch0 + ch) * 25 + r * 5 + s] * (float)nb[r][tk + s][ch];
        const float ctr = (float)nb[2][tk + 2][ch];
        A[tk][ch0 + ch] = (__bf16)(ctr + gelu_f(acc));
      }
      __syncthreads();
    }

    // fc2: (64 x 256) @ (256 x 64) + bias + residual, write to (b,c,h,w)
    const int nt = wvu;
    for (int mt = 0; mt < 4; ++mt) {
      v8f acc = {};
      for (int ks = 0; ks < 8; ++ks)
        acc = wmma_bf16(frag_a(&A[0][0], mt * 16, HDIM, ks * 32, lane),
                        frag_b(w2, nt * 16, HDIM, ks * 32, lane), acc);
      const int n = nt * 16 + col;
      const float bb = b2[n];
      const int rb = mt * 16 + rhalf;
      for (int j = 0; j < 8; ++j) {
        const int m = rb + j;
        const size_t tok = (size_t)bimg * HW + (size_t)y * IMGW + x0 + m;
        out[((size_t)bimg * CDIM + n) * HW + (size_t)y * IMGW + x0 + m] =
            x3[tok * CDIM + n] + acc[j] + bb;
      }
    }
    // next tile's first nb-load barrier protects A reuse
  }
}

extern "C" void kernel_launch(void* const* d_in, const int* in_sizes, int n_in,
                              void* d_out, int out_size, void* d_ws, size_t ws_size,
                              hipStream_t stream) {
  (void)in_sizes; (void)n_in; (void)out_size; (void)ws_size;
  const float* x      = (const float*)d_in[0];
  const float* n1w    = (const float*)d_in[1];
  const float* n1b    = (const float*)d_in[2];
  const float* qkvw   = (const float*)d_in[3];
  const float* qkvb   = (const float*)d_in[4];
  const float* rpb    = (const float*)d_in[5];
  const float* projw  = (const float*)d_in[6];
  const float* projb  = (const float*)d_in[7];
  const float* n2w    = (const float*)d_in[8];
  const float* n2b    = (const float*)d_in[9];
  const float* fc1w   = (const float*)d_in[10];
  const float* fc1b   = (const float*)d_in[11];
  const float* dww    = (const float*)d_in[12];
  const float* dwb    = (const float*)d_in[13];
  const float* fc2w   = (const float*)d_in[14];
  const float* fc2b   = (const float*)d_in[15];

  const size_t ntok = (size_t)BATCH * IMGH * IMGW;                       // 262144
  float*  x3   = (float*)d_ws;                                           // 67 MB
  __bf16* hidb = (__bf16*)((char*)d_ws + ntok * CDIM * sizeof(float));   // 134 MB
  __bf16* wbf  = (__bf16*)((char*)d_ws + ntok * CDIM * sizeof(float)
                                       + ntok * HDIM * sizeof(__bf16)); // 96 KB
  __bf16* qkvw_bf  = wbf;
  __bf16* projw_bf = wbf + 12288;
  __bf16* fc1w_bf  = wbf + 16384;
  __bf16* fc2w_bf  = wbf + 32768;

  wcvt_kernel<<<dim3(192), dim3(256), 0, stream>>>(qkvw, projw, fc1w, fc2w, wbf);
  attn_kernel<<<dim3(4096), dim3(128), 0, stream>>>(x, n1w, n1b, qkvw_bf, qkvb, rpb,
                                                    projw_bf, projb, x3);
  ffn1_kernel<<<dim3(1024), dim3(128), 0, stream>>>(x3, n2w, n2b, fc1w_bf, fc1b, hidb);
  ffn2_kernel<<<dim3(1024), dim3(128), 0, stream>>>(hidb, dww, dwb, fc2w_bf, fc2b, x3,
                                                    (float*)d_out);
}
